// BiLSTM_29394756174395
// MI455X (gfx1250) — matrix operand order repaired and checked by reading, not compiled, not run on hardware
//
#include <hip/hip_runtime.h>
#include <hip/hip_bf16.h>

typedef __bf16 bf16_t;
typedef __attribute__((ext_vector_type(16))) __bf16 v16bf;
typedef __attribute__((ext_vector_type(8)))  float  v8f;

#define TSTEPS 512
#define BATCH  32
#define DIN    256
#define HDIM   512
#define NWG_FW 32
#define HBUF   (BATCH*HDIM)

// dynamic-LDS slice layout (elements): [f0x 64xDIN][f0h 64xHDIM][f1x 64xHDIM][f1h 64xHDIM]
#define SLICE_ELEMS (64*DIN + 3*64*HDIM)   // 114688 elems = 229376 bytes

struct B32x2 { uint4 lo; uint4 hi; };

static __device__ inline v8f splat8(float v){
  v8f r;
#pragma unroll
  for (int i = 0; i < 8; ++i) r[i] = v;
  return r;
}

// A fragment: 16x32 bf16 (M x K), row-major, stride rstride (elements).
// lanes 0-15 -> M=lane, K = k0+{0..15}; lanes 16-31 K = k0+{16..31}.
static __device__ inline v16bf load_a(const bf16_t* __restrict__ base, size_t rstride, int m0, int k0){
  const int lane = threadIdx.x & 31;
  const int half = lane >> 4;
  const bf16_t* p = base + (size_t)(m0 + (lane & 15)) * rstride + (size_t)(k0 + half * 16);
  B32x2 u;
  u.lo = *(const uint4*)(p);
  u.hi = *(const uint4*)(p + 8);
  return __builtin_bit_cast(v16bf, u);
}

// B fragment from W^T [N][K] row-major in GLOBAL memory (bw path).
static __device__ inline v16bf load_b_glb(const bf16_t* __restrict__ wt, int ld, int n0, int k0){
  const int lane = threadIdx.x & 31;
  const int half = lane >> 4;
  const bf16_t* p = wt + (size_t)(n0 + (lane & 15)) * ld + (size_t)(k0 + half * 16);
  B32x2 u;
  u.lo = *(const uint4*)(p);
  u.hi = *(const uint4*)(p + 8);
  return __builtin_bit_cast(v16bf, u);
}

// B fragment from an LDS-resident slice: 64 rows (g*16 + c), ld = K.
static __device__ inline v16bf load_b_lds(const bf16_t* sbase, int K, int g, int k0){
  const int lane = threadIdx.x & 31;
  const int half = lane >> 4;
  const bf16_t* p = sbase + (size_t)(g * 16 + (lane & 15)) * K + (size_t)(k0 + half * 16);
  B32x2 u;
  u.lo = *(const uint4*)(p);
  u.hi = *(const uint4*)(p + 8);
  return __builtin_bit_cast(v16bf, u);
}

// Async-stage one weight slice (64 z-rows x K) into LDS via the CDNA5 async LDS path.
// Cooperative across all 64 threads of the WG; tracked by ASYNCcnt.
static __device__ inline void stage_slice(const bf16_t* __restrict__ src, int K, int cols16,
                                          bf16_t* dstLds){
  const int tid = threadIdx.x;               // 0..63
  const int chunks = K >> 3;                 // 16-byte chunks per row (32 or 64)
  const int total = 64 * chunks;
  for (int i = tid; i < total; i += 64){
    const int s  = i / chunks;               // slice row 0..63
    const int ch = i - s * chunks;
    const int g = s >> 4, c = s & 15;
    const bf16_t* sp = src + (size_t)(g * HDIM + cols16 + c) * K + (size_t)ch * 8;
    bf16_t*       dp = dstLds + (size_t)s * K + (size_t)ch * 8;
    unsigned long long ga = (unsigned long long)(uintptr_t)sp;
    unsigned int       la = (unsigned int)(uintptr_t)dp;
    asm volatile("global_load_async_to_lds_b128 %0, %1, off"
                 :: "v"(la), "v"(ga) : "memory");
  }
}

// 4 gate tiles (one M row-tile) accumulated from LDS-resident B.
static __device__ inline void mma4_lds(v8f* acc, const bf16_t* __restrict__ abase, size_t arstride,
                                       int m0, const bf16_t* sb, int K){
  for (int k0 = 0; k0 < K; k0 += 32){
    v16bf a = load_a(abase, arstride, m0, k0);
#pragma unroll
    for (int g = 0; g < 4; ++g){
      v16bf b = load_b_lds(sb, K, g, k0);
      acc[g] = __builtin_amdgcn_wmma_f32_16x16x32_bf16(false, a, false, b, (short)0, acc[g], false, false);
    }
  }
}

// 4 gate tiles accumulated from global B (bw path, weights read exactly once).
static __device__ inline void mma4_glb(v8f* acc, const bf16_t* __restrict__ abase, size_t arstride,
                                       int m0, const bf16_t* __restrict__ wt, int kld, int cols16, int K){
  for (int k0 = 0; k0 < K; k0 += 32){
    v16bf a = load_a(abase, arstride, m0, k0);
    if (k0 + 32 < K)
      __builtin_prefetch((const void*)(wt + (size_t)cols16 * kld + (size_t)(k0 + 32)), 0, 1);
#pragma unroll
    for (int g = 0; g < 4; ++g){
      v16bf b = load_b_glb(wt, kld, g * HDIM + cols16, k0);
      acc[g] = __builtin_amdgcn_wmma_f32_16x16x32_bf16(false, a, false, b, (short)0, acc[g], false, false);
    }
  }
}

static __device__ inline void init4(v8f* acc, const float* __restrict__ bias, int cols16){
  const int n = cols16 + (threadIdx.x & 15);
#pragma unroll
  for (int g = 0; g < 4; ++g)
    acc[g] = splat8(bias[g * HDIM + n]);
}

// LSTM cell pointwise update for one 16x16 row-tile, in registers.
// C/D layout: element (m,n) -> reg m&7, lane 16*(m>=8)+n, rows offset by m0.
// lv: per-lane preloaded l[mrow] (index r); may be nullptr when !use_mask && out_always.
static __device__ inline void cell4(v8f* acc, v8f& creg, v8f& hreg,
                                    const int* lv, int t, bool use_mask, int m0,
                                    bf16_t* __restrict__ hstate, bf16_t* __restrict__ hout,
                                    int cols16, float* __restrict__ outp, int outoff, bool out_always){
  const int lane = threadIdx.x & 31;
  const int half = lane >> 4;
  const int n = cols16 + (lane & 15);
#pragma unroll
  for (int r = 0; r < 8; ++r){
    const int mrow = m0 + r + 8 * half;
    float si = 1.0f / (1.0f + __expf(-acc[0][r]));   // input gate
    float tg = tanhf(acc[1][r]);                     // cell candidate
    float sf = 1.0f / (1.0f + __expf(-acc[2][r]));   // forget gate
    float so = 1.0f / (1.0f + __expf(-acc[3][r]));   // output gate
    float cn = sf * creg[r] + si * tg;
    float hn = so * tanhf(cn);
    bool msk = (!use_mask) || (t < lv[r]);
    float cm = msk ? cn : creg[r];
    float hm = msk ? hn : hreg[r];
    creg[r] = cm;
    hreg[r] = hm;
    if (hstate) hstate[mrow * HDIM + n] = (bf16_t)hm;   // masked state -> recurrence
    if (hout)   hout[mrow * HDIM + n]   = (bf16_t)hn;   // unmasked -> next layer input
    if (outp){
      if (out_always || (t == lv[r] - 1))
        outp[mrow * (2 * HDIM) + outoff + n] = hn;
    }
  }
}

// Global split-barrier across the 32 persistent forward workgroups (2 waves each).
static __device__ inline void gbar(int* cnt, int& epoch){
  __threadfence();
  __syncthreads();
  if (threadIdx.x == 0){
    __hip_atomic_fetch_add(cnt, 1, __ATOMIC_RELEASE, __HIP_MEMORY_SCOPE_AGENT);
    const int target = (epoch + 1) * NWG_FW;
    while (__hip_atomic_load(cnt, __ATOMIC_ACQUIRE, __HIP_MEMORY_SCOPE_AGENT) < target)
      __builtin_amdgcn_s_sleep(1);
  }
  __syncthreads();
  __threadfence();
  ++epoch;
}

// ---------------- prep kernels ----------------

__global__ void k_init(int* bar, bf16_t* Hst0, bf16_t* Hst1){
  int i = blockIdx.x * blockDim.x + threadIdx.x;
  if (i == 0) *bar = 0;
  if (i < HBUF){ Hst0[i] = (bf16_t)0.0f; Hst1[i] = (bf16_t)0.0f; }
}

__global__ void k_cvt_bf16(const float* __restrict__ src, bf16_t* __restrict__ dst, int n){
  int i = blockIdx.x * blockDim.x + threadIdx.x;
  if (i < n) dst[i] = (bf16_t)src[i];
}

__global__ void k_xlast(const float* __restrict__ x, const int* __restrict__ l, bf16_t* __restrict__ xl){
  int b = blockIdx.x, k = threadIdx.x;
  int t = l[b] - 1;
  xl[b * DIN + k] = (bf16_t)x[((size_t)b * TSTEPS + t) * DIN + k];
}

// W [K][2048] row-major -> W^T [2048][K] bf16 row-major (B-fragments become contiguous).
__global__ void k_tr(const float* __restrict__ w, bf16_t* __restrict__ wt, int K){
  int i = blockIdx.x * blockDim.x + threadIdx.x;
  if (i < K * 2048){
    int nn = i / K, kk = i - nn * K;
    wt[i] = (bf16_t)w[(size_t)kk * 2048 + nn];
  }
}

// ---------------- forward chain: 32 persistent WGs (2 waves: M-split), LDS-resident weights ----------------

__global__ void __launch_bounds__(64) fw_kernel(
    const int* __restrict__ l,
    const float* __restrict__ b_f0, const float* __restrict__ b_f1,
    const bf16_t* __restrict__ Xbf,
    const bf16_t* __restrict__ Wf0x, const bf16_t* __restrict__ Wf0h,
    const bf16_t* __restrict__ Wf1x, const bf16_t* __restrict__ Wf1h,
    bf16_t* __restrict__ Hst0, bf16_t* __restrict__ Hst1,
    bf16_t* __restrict__ Hout0,
    int* bar, float* __restrict__ out)
{
  extern __shared__ bf16_t sw[];
  const int cols16 = blockIdx.x * 16;          // this WG's 16 H columns
  const int wave = threadIdx.x >> 5;           // M-split: wave0 rows 0-15, wave1 rows 16-31
  const int m0 = wave * 16;
  const int half = (threadIdx.x & 31) >> 4;

  bf16_t* s_f0x = sw;
  bf16_t* s_f0h = sw + 64*DIN;
  bf16_t* s_f1x = sw + 64*DIN + 64*HDIM;
  bf16_t* s_f1h = sw + 64*DIN + 2*64*HDIM;

  // one-time async fill of the 224KB weight slice (ASYNCcnt-tracked DMA into LDS)
  stage_slice(Wf0x, DIN,  cols16, s_f0x);
  stage_slice(Wf0h, HDIM, cols16, s_f0h);
  stage_slice(Wf1x, HDIM, cols16, s_f1x);
  stage_slice(Wf1h, HDIM, cols16, s_f1h);
  asm volatile("s_wait_asynccnt 0" ::: "memory");
  __syncthreads();

  int lv[8];
#pragma unroll
  for (int r = 0; r < 8; ++r)
    lv[r] = l[m0 + r + 8*half];

  v8f c0 = splat8(0.0f), h0 = splat8(0.0f);
  v8f c1 = splat8(0.0f), h1 = splat8(0.0f);
  int epoch = 0, p = 0;

  for (int t = 0; t < TSTEPS; ++t){
    // layer 0: z = x_t @ Wx0 + h0 @ Wh0 + b0
    v8f acc[4];
    init4(acc, b_f0, cols16);
    mma4_lds(acc, Xbf + (size_t)t * DIN, (size_t)TSTEPS * DIN, m0, s_f0x, DIN);
    mma4_lds(acc, Hst0 + (size_t)p * HBUF, HDIM,               m0, s_f0h, HDIM);
    cell4(acc, c0, h0, lv, t, true, m0,
          Hst0 + (size_t)(1 - p) * HBUF, Hout0, cols16, nullptr, 0, false);
    gbar(bar, epoch);
    // layer 1: z = h0_new @ Wx1 + h1 @ Wh1 + b1
    init4(acc, b_f1, cols16);
    mma4_lds(acc, Hout0, HDIM,                   m0, s_f1x, HDIM);
    mma4_lds(acc, Hst1 + (size_t)p * HBUF, HDIM, m0, s_f1h, HDIM);
    cell4(acc, c1, h1, lv, t, true, m0,
          Hst1 + (size_t)(1 - p) * HBUF, nullptr, cols16, out, 0, false);
    gbar(bar, epoch);
    p ^= 1;
  }
}

// ---------------- backward direction: exactly one LSTM step on x[b, l[b]-1] ----------------

__global__ void __launch_bounds__(256) bw_kernel(
    const float* __restrict__ b_b0, const float* __restrict__ b_b1,
    const bf16_t* __restrict__ Xlast,
    const bf16_t* __restrict__ Wb0x, const bf16_t* __restrict__ Wb1x,
    bf16_t* __restrict__ bwH0, float* __restrict__ out)
{
  const int wave = threadIdx.x >> 5;      // 8 waves: 2 M-tiles x 4 column groups
  const int m0 = (wave & 1) * 16;
  const int cb = (wave >> 1) * 8;         // 8 column-tiles per group
  for (int cg = 0; cg < 8; ++cg){
    const int cols16 = (cb + cg) * 16;
    v8f acc[4]; init4(acc, b_b0, cols16);
    v8f cz = splat8(0.0f), hz = splat8(0.0f);
    mma4_glb(acc, Xlast, DIN, m0, Wb0x, DIN, cols16, DIN);
    cell4(acc, cz, hz, nullptr, 0, false, m0, nullptr, bwH0, cols16, nullptr, 0, false);
  }
  __threadfence();
  __syncthreads();
  for (int cg = 0; cg < 8; ++cg){
    const int cols16 = (cb + cg) * 16;
    v8f acc[4]; init4(acc, b_b1, cols16);
    v8f cz = splat8(0.0f), hz = splat8(0.0f);
    mma4_glb(acc, bwH0, HDIM, m0, Wb1x, HDIM, cols16, HDIM);
    cell4(acc, cz, hz, nullptr, 0, false, m0, nullptr, nullptr, cols16, out, HDIM, true);
  }
}

// ---------------- host launcher ----------------

extern "C" void kernel_launch(void* const* d_in, const int* in_sizes, int n_in,
                              void* d_out, int out_size, void* d_ws, size_t ws_size,
                              hipStream_t stream)
{
  const float* x    = (const float*)d_in[0];
  const int*   l    = (const int*)  d_in[1];
  const float* f0Wx = (const float*)d_in[2];
  const float* f0Wh = (const float*)d_in[3];
  const float* f0b  = (const float*)d_in[4];
  const float* f1Wx = (const float*)d_in[5];
  const float* f1Wh = (const float*)d_in[6];
  const float* f1b  = (const float*)d_in[7];
  const float* b0Wx = (const float*)d_in[8];
  const float* b0b  = (const float*)d_in[10];
  const float* b1Wx = (const float*)d_in[11];
  const float* b1b  = (const float*)d_in[13];
  float* out = (float*)d_out;

  char* ws = (char*)d_ws;
  size_t off = 0;
  auto take = [&](size_t bytes) -> char* {
    char* p = ws + off;
    off += (bytes + 255) & ~(size_t)255;
    return p;
  };
  int*    bar   = (int*)   take(256);
  bf16_t* Xbf   = (bf16_t*)take((size_t)BATCH * TSTEPS * DIN * 2);
  bf16_t* Xlast = (bf16_t*)take((size_t)BATCH * DIN * 2);
  bf16_t* Wf0x  = (bf16_t*)take((size_t)2048 * DIN  * 2);
  bf16_t* Wf0h  = (bf16_t*)take((size_t)2048 * HDIM * 2);
  bf16_t* Wf1x  = (bf16_t*)take((size_t)2048 * HDIM * 2);
  bf16_t* Wf1h  = (bf16_t*)take((size_t)2048 * HDIM * 2);
  bf16_t* Wb0x  = (bf16_t*)take((size_t)2048 * DIN  * 2);
  bf16_t* Wb1x  = (bf16_t*)take((size_t)2048 * HDIM * 2);
  bf16_t* Hst0  = (bf16_t*)take((size_t)2 * HBUF * 2);
  bf16_t* Hst1  = (bf16_t*)take((size_t)2 * HBUF * 2);
  bf16_t* Hout0 = (bf16_t*)take((size_t)HBUF * 2);
  bf16_t* bwH0  = (bf16_t*)take((size_t)HBUF * 2);

  k_init<<<dim3((HBUF + 255) / 256), dim3(256), 0, stream>>>(bar, Hst0, Hst1);
  {
    int n = BATCH * TSTEPS * DIN;
    k_cvt_bf16<<<dim3((n + 255) / 256), dim3(256), 0, stream>>>(x, Xbf, n);
  }
  k_xlast<<<dim3(BATCH), dim3(DIN), 0, stream>>>(x, l, Xlast);
  k_tr<<<dim3((2048 * DIN  + 255) / 256), dim3(256), 0, stream>>>(f0Wx, Wf0x, DIN);
  k_tr<<<dim3((2048 * HDIM + 255) / 256), dim3(256), 0, stream>>>(f0Wh, Wf0h, HDIM);
  k_tr<<<dim3((2048 * HDIM + 255) / 256), dim3(256), 0, stream>>>(f1Wx, Wf1x, HDIM);
  k_tr<<<dim3((2048 * HDIM + 255) / 256), dim3(256), 0, stream>>>(f1Wh, Wf1h, HDIM);
  k_tr<<<dim3((2048 * DIN  + 255) / 256), dim3(256), 0, stream>>>(b0Wx, Wb0x, DIN);
  k_tr<<<dim3((2048 * HDIM + 255) / 256), dim3(256), 0, stream>>>(b1Wx, Wb1x, HDIM);

  const size_t fwLds = (size_t)SLICE_ELEMS * 2;   // 229376 bytes
  (void)hipFuncSetAttribute((const void*)fw_kernel,
                            hipFuncAttributeMaxDynamicSharedMemorySize, (int)fwLds);

  bw_kernel<<<dim3(1), dim3(256), 0, stream>>>(b0b, b1b, Xlast, Wb0x, Wb1x, bwH0, out);
  fw_kernel<<<dim3(NWG_FW), dim3(64), fwLds, stream>>>(
      l, f0b, f1b, Xbf, Wf0x, Wf0h, Wf1x, Wf1h, Hst0, Hst1, Hout0, bar, out);

  (void)in_sizes; (void)n_in; (void)out_size; (void)ws_size;
}
